// CABlock_20220706030153
// MI455X (gfx1250) — matrix operand order repaired and checked by reading, not compiled, not run on hardware
//
#include <hip/hip_runtime.h>
#include <hip/hip_bf16.h>

// ---------------------------------------------------------------------------
// CDNA5 (gfx1250) cross-attention block.
// f16 storage / f32 WMMA accumulation (v_wmma_f32_16x16x32_f16), wave32.
// V tiles staged global->LDS via the Tensor Data Mover (tensor_load_to_lds,
// TENSORcnt), double-buffered so the DMA overlaps PV + next QK^T/softmax.
// Fragment layouts used here (contiguous K order, same for A and B):
//   A (16x32 f16): lane l -> row M=l&15, half=l>>4;
//                  elem e -> K = half*16 + e                    (two 16B loads)
//   B (32x16 f16): lane l -> col N=l&15, half=l>>4;
//                  elem e -> K = half*16 + e                    (one 32B load)
//   C/D (16x16 f32): lane l, vgpr r -> (M = r + 8*(l>>4), N = l&15)
// ---------------------------------------------------------------------------

typedef _Float16 half_t;
typedef __attribute__((ext_vector_type(16))) _Float16 v16h;
typedef __attribute__((ext_vector_type(8)))  _Float16 v8h;
typedef __attribute__((ext_vector_type(8)))  float    v8f;
typedef __attribute__((ext_vector_type(4)))  unsigned int u32x4;
typedef __attribute__((ext_vector_type(8)))  unsigned int u32x8;

static __device__ __forceinline__ v8f wmma_f16(v16h a, v16h b, v8f c) {
  return __builtin_amdgcn_wmma_f32_16x16x32_f16(false, a, false, b, (short)0, c,
                                                false, false);
}

// A fragment from row-major f16 [*, lda], tile origin (mbase, kbase).
static __device__ __forceinline__ v16h load_a_frag(const half_t* __restrict__ A,
                                                   int lda, int mbase, int kbase) {
  const int lane = threadIdx.x & 31;
  const int m = mbase + (lane & 15);
  const int hh = lane >> 4;
  const half_t* p = A + (size_t)m * lda + kbase + hh * 16;
  v8h lo = *(const v8h*)(p);        // K = hh*16 + 0..7
  v8h hi = *(const v8h*)(p + 8);    // K = hh*16 + 8..15
  v16h r;
#pragma unroll
  for (int i = 0; i < 8; ++i) { r[i] = lo[i]; r[i + 8] = hi[i]; }
  return r;
}

// B fragment where the K x N operand is stored TRANSPOSED row-major as Bt[N][K]
// (weights stored as W^T, or K-matrix for Q*K^T). Contiguous 32B per lane.
static __device__ __forceinline__ v16h load_bt_frag(const half_t* __restrict__ Bt,
                                                    int ldb, int nbase, int kbase) {
  const int lane = threadIdx.x & 31;
  const int n = nbase + (lane & 15);
  const int hh = lane >> 4;
  return *(const v16h*)(Bt + (size_t)n * ldb + kbase + hh * 16);
}

// B fragment from a row-major K x N tile (V tile in LDS).
static __device__ __forceinline__ v16h load_b_rowmajor(const half_t* B, int ldb,
                                                       int kbase, int nbase) {
  const int lane = threadIdx.x & 31;
  const int n = nbase + (lane & 15);
  const int hh = lane >> 4;
  const half_t* p = B + (size_t)(kbase + hh * 16) * ldb + n;
  v16h r;
#pragma unroll
  for (int e = 0; e < 16; ++e) r[e] = p[(size_t)e * ldb];
  return r;
}

// ---------------------------------------------------------------------------
// Tensor Data Mover: stage a [rows x 64] f16 tile (row stride `row_stride`
// elems) from global into LDS (compacted to row stride 64).  D# per
// CDNA5 ISA.
// ---------------------------------------------------------------------------
static __device__ __forceinline__ void tdm_load_tile_64w(const half_t* gsrc,
                                                         unsigned lds_addr,
                                                         int rows, int row_stride) {
  const unsigned long long ga = (unsigned long long)(uintptr_t)gsrc;
  u32x4 g0;
  g0[0] = 1u;                                   // count=1, user mode
  g0[1] = lds_addr;                             // LDS byte address
  g0[2] = (unsigned)(ga & 0xffffffffu);         // global_addr[31:0]
  g0[3] = (unsigned)((ga >> 32) & 0x01ffffffu) | (2u << 30);  // [56:32] | type=2
  u32x8 g1;
  g1[0] = (1u << 16);                           // wg_mask=0, data_size=1 (2B)
  g1[1] = (64u << 16);                          // tensor_dim0 = 64  (bits 63:48)
  g1[2] = ((unsigned)rows << 16);               // tensor_dim1 lo    (bits 95:80)
  g1[3] = (64u << 16);                          // tile_dim0 = 64    (bits 127:112)
  g1[4] = (unsigned)rows;                       // tile_dim1         (bits 143:128)
  g1[5] = (unsigned)row_stride;                 // tensor_dim0_stride lo32
  g1[6] = 0u;
  g1[7] = 0u;
  asm volatile("tensor_load_to_lds %0, %1" :: "s"(g0), "s"(g1) : "memory");
}

// ---------------------------------------------------------------------------
// Weight convert + transpose: W[K][N] f32 -> Wt[N][K] f16
// ---------------------------------------------------------------------------
__global__ void __launch_bounds__(256) cvt_wt_kernel(const float* __restrict__ W,
                                                     half_t* __restrict__ Wt,
                                                     int K, int N) {
  int idx = blockIdx.x * 256 + threadIdx.x;
  if (idx < K * N) {
    int k = idx / N, n = idx - k * N;
    Wt[(size_t)n * K + k] = (half_t)__builtin_nontemporal_load(&W[idx]);
  }
}

// ---------------------------------------------------------------------------
// LayerNorm over channels of x[b, c, n] (channel-strided). One block per token.
// ---------------------------------------------------------------------------
__global__ void __launch_bounds__(256) ln_x_kernel(const float* __restrict__ x,
                                                   const float* __restrict__ g,
                                                   const float* __restrict__ bt,
                                                   float* __restrict__ xn32,
                                                   half_t* __restrict__ xn16,
                                                   int C, int NTOK) {
  const int row = blockIdx.x;              // b*NTOK + n
  const int b = row / NTOK, n = row - b * NTOK;
  const int c = threadIdx.x;
  float v = __builtin_nontemporal_load(&x[((size_t)b * C + c) * NTOK + n]);
  float s = v, s2 = v * v;
#pragma unroll
  for (int d = 16; d >= 1; d >>= 1) {
    s  += __shfl_xor(s,  d, 32);
    s2 += __shfl_xor(s2, d, 32);
  }
  __shared__ float rs[8], rs2[8];
  const int w = threadIdx.x >> 5;
  if ((threadIdx.x & 31) == 0) { rs[w] = s; rs2[w] = s2; }
  __syncthreads();
  float ts = 0.f, ts2 = 0.f;
#pragma unroll
  for (int i = 0; i < 8; ++i) { ts += rs[i]; ts2 += rs2[i]; }
  const float mu   = ts / (float)C;
  const float var  = ts2 / (float)C - mu * mu;
  const float rstd = rsqrtf(var + 1e-5f);
  const float o = (v - mu) * rstd * g[c] + bt[c];
  xn32[(size_t)row * C + c] = o;
  xn16[(size_t)row * C + c] = (half_t)o;
}

__global__ void __launch_bounds__(256) ln_y_kernel(const float* __restrict__ y,
                                                   const float* __restrict__ g,
                                                   const float* __restrict__ bt,
                                                   half_t* __restrict__ yn16, int C) {
  const int row = blockIdx.x;
  const int c = threadIdx.x;
  float v = __builtin_nontemporal_load(&y[(size_t)row * C + c]);
  float s = v, s2 = v * v;
#pragma unroll
  for (int d = 16; d >= 1; d >>= 1) {
    s  += __shfl_xor(s,  d, 32);
    s2 += __shfl_xor(s2, d, 32);
  }
  __shared__ float rs[8], rs2[8];
  const int w = threadIdx.x >> 5;
  if ((threadIdx.x & 31) == 0) { rs[w] = s; rs2[w] = s2; }
  __syncthreads();
  float ts = 0.f, ts2 = 0.f;
#pragma unroll
  for (int i = 0; i < 8; ++i) { ts += rs[i]; ts2 += rs2[i]; }
  const float mu   = ts / (float)C;
  const float var  = ts2 / (float)C - mu * mu;
  const float rstd = rsqrtf(var + 1e-5f);
  yn16[(size_t)row * C + c] = (half_t)((v - mu) * rstd * g[c] + bt[c]);
}

// ---------------------------------------------------------------------------
// WMMA GEMM: C[M,N] = A[M,K] @ W[K,N] (+bias +resid), W stored transposed.
// 4 waves/block; each wave owns a 32x64 tile (2 A-frags x 4 B-frags ->
// 8 independent WMMAs per K-step for load/compute overlap).
// ---------------------------------------------------------------------------
__global__ void __launch_bounds__(128) gemm_nt_kernel(
    const half_t* __restrict__ A, const half_t* __restrict__ Bt,
    half_t* __restrict__ out16, float* __restrict__ out32,
    const float* __restrict__ bias, const float* __restrict__ resid,
    int M, int N, int K) {
  const int w = threadIdx.x >> 5;
  const int lane = threadIdx.x & 31;
  const int hh = lane >> 4;
  const int mb = blockIdx.y * 128 + w * 32;
  const int nb = blockIdx.x * 64;

  v8f acc[2][4];
#pragma unroll
  for (int i = 0; i < 2; ++i)
#pragma unroll
    for (int j = 0; j < 4; ++j) acc[i][j] = {};

  for (int kk = 0; kk < K; kk += 32) {
    const v16h a0 = load_a_frag(A, K, mb, kk);
    const v16h a1 = load_a_frag(A, K, mb + 16, kk);
    const v16h b0 = load_bt_frag(Bt, K, nb,      kk);
    const v16h b1 = load_bt_frag(Bt, K, nb + 16, kk);
    const v16h b2 = load_bt_frag(Bt, K, nb + 32, kk);
    const v16h b3 = load_bt_frag(Bt, K, nb + 48, kk);
    if (kk + 32 < K) {  // gfx1250: near-cache prefetch of the next A chunk
      __builtin_prefetch(A + (size_t)(mb + (lane & 15)) * K + kk + 32, 0, 3);
      __builtin_prefetch(A + (size_t)(mb + 16 + (lane & 15)) * K + kk + 32, 0, 3);
    }
    acc[0][0] = wmma_f16(a0, b0, acc[0][0]);
    acc[0][1] = wmma_f16(a0, b1, acc[0][1]);
    acc[0][2] = wmma_f16(a0, b2, acc[0][2]);
    acc[0][3] = wmma_f16(a0, b3, acc[0][3]);
    acc[1][0] = wmma_f16(a1, b0, acc[1][0]);
    acc[1][1] = wmma_f16(a1, b1, acc[1][1]);
    acc[1][2] = wmma_f16(a1, b2, acc[1][2]);
    acc[1][3] = wmma_f16(a1, b3, acc[1][3]);
  }

#pragma unroll
  for (int mt = 0; mt < 2; ++mt)
#pragma unroll
    for (int nt = 0; nt < 4; ++nt) {
      const int col = nb + nt * 16 + (lane & 15);
      const float bcol = bias ? bias[col] : 0.0f;
#pragma unroll
      for (int r = 0; r < 8; ++r) {
        const int row = mb + mt * 16 + r + 8 * hh;
        float v = acc[mt][nt][r] + bcol;
        if (resid) v += resid[(size_t)row * N + col];
        if (out32) __builtin_nontemporal_store(v, &out32[(size_t)row * N + col]);
        else       out16[(size_t)row * N + col] = (half_t)v;
      }
    }
}

// ---------------------------------------------------------------------------
// Flash attention: 256 threads / 8 waves per block; each wave owns 16 queries
// x 64 head-dim (128 queries per block halves K/V L2 re-reads).  Online
// softmax over key tiles of 32.  V tiles double-buffered in LDS via the TDM:
// the DMA for tile i+1 overlaps PV(i) and S/softmax(i+1).  One barrier/tile.
// ---------------------------------------------------------------------------
__global__ void __launch_bounds__(256) attn_kernel(
    const half_t* __restrict__ Q, const half_t* __restrict__ Kc,
    const half_t* __restrict__ V, half_t* __restrict__ O,
    int n_q, int n_kv) {
  __shared__ __align__(64) half_t Vlds[2][32 * 64];   // TDM-staged V tiles
  __shared__ __align__(64) half_t Plds[8][16 * 32];   // per-wave P staging tile
  const int w = threadIdx.x >> 5;
  const int lane = threadIdx.x & 31;
  const int hh = lane >> 4;
  const int bh = blockIdx.y;                 // b*8 + h
  const int b = bh >> 3, h = bh & 7;
  const int qbase = blockIdx.x * 128 + w * 16;

  const half_t* qp = Q  + (size_t)b * n_q  * 512 + h * 64;
  const half_t* kp = Kc + (size_t)b * n_kv * 512 + h * 64;
  const half_t* vp = V  + (size_t)b * n_kv * 512 + h * 64;

  const v16h aQ0 = load_a_frag(qp, 512, qbase, 0);
  const v16h aQ1 = load_a_frag(qp, 512, qbase, 32);

  v8f o0 = {}, o1 = {}, o2 = {}, o3 = {};
  float mrow[8], lrow[8];
#pragma unroll
  for (int r = 0; r < 8; ++r) { mrow[r] = -1e30f; lrow[r] = 0.f; }

  half_t* pw = &Plds[w][0];

  // Preload first V tile.
  if (w == 0)
    tdm_load_tile_64w(vp, (unsigned)(uintptr_t)&Vlds[0][0], 32, 512);

  for (int kt = 0, it = 0; kt < n_kv; kt += 32, ++it) {
    const int cur = it & 1;

    // S = Q @ K^T (K row-major [key][d] acts as the transposed B operand).
    v8f s0 = {}, s1 = {};
    s0 = wmma_f16(aQ0, load_bt_frag(kp, 512, kt,      0),  s0);
    s0 = wmma_f16(aQ1, load_bt_frag(kp, 512, kt,      32), s0);
    s1 = wmma_f16(aQ0, load_bt_frag(kp, 512, kt + 16, 0),  s1);
    s1 = wmma_f16(aQ1, load_bt_frag(kp, 512, kt + 16, 32), s1);

    // Online softmax; row reduction = xor-shuffle within the 16-lane half.
#pragma unroll
    for (int r = 0; r < 8; ++r) {
      float v0 = s0[r] * 0.125f;           // dim_head^-0.5
      float v1 = s1[r] * 0.125f;
      float mx = fmaxf(v0, v1);
#pragma unroll
      for (int d = 1; d < 16; d <<= 1) mx = fmaxf(mx, __shfl_xor(mx, d, 32));
      const float mnew  = fmaxf(mrow[r], mx);
      const float scale = __expf(mrow[r] - mnew);
      const float p0 = __expf(v0 - mnew);
      const float p1 = __expf(v1 - mnew);
      float ps = p0 + p1;
#pragma unroll
      for (int d = 1; d < 16; d <<= 1) ps += __shfl_xor(ps, d, 32);
      lrow[r] = lrow[r] * scale + ps;
      mrow[r] = mnew;
      o0[r] *= scale; o1[r] *= scale; o2[r] *= scale; o3[r] *= scale;
      const int prow = r + 8 * hh;         // D-layout -> LDS row-major P tile
      pw[prow * 32 +      (lane & 15)] = (half_t)p0;
      pw[prow * 32 + 16 + (lane & 15)] = (half_t)p1;
    }

    // Publish V tile `cur`, then immediately kick the DMA for the next tile
    // into the other buffer (overlaps PV and the next S/softmax).
    if (w == 0) __builtin_amdgcn_s_wait_tensorcnt(0);
    __syncthreads();
    if (w == 0 && kt + 32 < n_kv)
      tdm_load_tile_64w(vp + (size_t)(kt + 32) * 512,
                        (unsigned)(uintptr_t)&Vlds[cur ^ 1][0], 32, 512);

    const v16h aP = load_a_frag(pw, 32, 0, 0);   // P as A operand (16x32)
    o0 = wmma_f16(aP, load_b_rowmajor(&Vlds[cur][0], 64, 0, 0),  o0);
    o1 = wmma_f16(aP, load_b_rowmajor(&Vlds[cur][0], 64, 0, 16), o1);
    o2 = wmma_f16(aP, load_b_rowmajor(&Vlds[cur][0], 64, 0, 32), o2);
    o3 = wmma_f16(aP, load_b_rowmajor(&Vlds[cur][0], 64, 0, 48), o3);
    // No trailing barrier: P is wave-private (in-order LDS), and buffer
    // cur^1 was last read in the previous iteration, before this tile's barrier.
  }

  half_t* op = O + (size_t)b * n_q * 512 + h * 64;
  v8f os[4] = {o0, o1, o2, o3};
#pragma unroll
  for (int nt = 0; nt < 4; ++nt)
#pragma unroll
    for (int r = 0; r < 8; ++r) {
      const int row = qbase + r + 8 * hh;
      op[(size_t)row * 512 + nt * 16 + (lane & 15)] = (half_t)(os[nt][r] / lrow[r]);
    }
}

// ---------------------------------------------------------------------------
// Host launcher
// ---------------------------------------------------------------------------
extern "C" void kernel_launch(void* const* d_in, const int* in_sizes, int n_in,
                              void* d_out, int out_size, void* d_ws, size_t ws_size,
                              hipStream_t stream) {
  (void)in_sizes; (void)n_in; (void)out_size; (void)ws_size;
  const float* x      = (const float*)d_in[0];
  const float* y      = (const float*)d_in[1];
  const float* ln_x_g = (const float*)d_in[2];
  const float* ln_x_b = (const float*)d_in[3];
  const float* ln_y_g = (const float*)d_in[4];
  const float* ln_y_b = (const float*)d_in[5];
  const float* Wq     = (const float*)d_in[6];
  const float* Wk     = (const float*)d_in[7];
  const float* Wv     = (const float*)d_in[8];
  const float* bv     = (const float*)d_in[9];
  const float* Wo     = (const float*)d_in[10];
  const float* bo     = (const float*)d_in[11];
  float* out = (float*)d_out;

  constexpr int B = 4, C = 256, NTOK = 4096, MTOK = 1024, INNER = 512;

  char* ws = (char*)d_ws;
  size_t off = 0;
  auto carve = [&](size_t bytes) -> void* {
    void* p = ws + off;
    off += (bytes + 255) & ~(size_t)255;
    return p;
  };
  float*  xn32 = (float*) carve((size_t)B * NTOK * C * sizeof(float));
  half_t* xn16 = (half_t*)carve((size_t)B * NTOK * C * sizeof(half_t));
  half_t* yn16 = (half_t*)carve((size_t)B * MTOK * C * sizeof(half_t));
  half_t* wq16 = (half_t*)carve((size_t)C * INNER * sizeof(half_t));   // [INNER][C]
  half_t* wk16 = (half_t*)carve((size_t)C * INNER * sizeof(half_t));
  half_t* wv16 = (half_t*)carve((size_t)C * INNER * sizeof(half_t));
  half_t* wo16 = (half_t*)carve((size_t)INNER * C * sizeof(half_t));   // [C][INNER]
  half_t* q16  = (half_t*)carve((size_t)B * NTOK * INNER * sizeof(half_t));
  half_t* k16  = (half_t*)carve((size_t)B * MTOK * INNER * sizeof(half_t));
  half_t* v16  = (half_t*)carve((size_t)B * MTOK * INNER * sizeof(half_t));
  half_t* at16 = (half_t*)carve((size_t)B * NTOK * INNER * sizeof(half_t));

  {
    int n1 = C * INNER;
    cvt_wt_kernel<<<(n1 + 255) / 256, 256, 0, stream>>>(Wq, wq16, C, INNER);
    cvt_wt_kernel<<<(n1 + 255) / 256, 256, 0, stream>>>(Wk, wk16, C, INNER);
    cvt_wt_kernel<<<(n1 + 255) / 256, 256, 0, stream>>>(Wv, wv16, C, INNER);
    cvt_wt_kernel<<<(n1 + 255) / 256, 256, 0, stream>>>(Wo, wo16, INNER, C);
  }
  ln_x_kernel<<<B * NTOK, 256, 0, stream>>>(x, ln_x_g, ln_x_b, xn32, xn16, C, NTOK);
  ln_y_kernel<<<B * MTOK, 256, 0, stream>>>(y, ln_y_g, ln_y_b, yn16, C);

  gemm_nt_kernel<<<dim3(INNER / 64, (B * NTOK) / 128), 128, 0, stream>>>(
      xn16, wq16, q16, nullptr, nullptr, nullptr, B * NTOK, INNER, C);
  gemm_nt_kernel<<<dim3(INNER / 64, (B * MTOK) / 128), 128, 0, stream>>>(
      yn16, wk16, k16, nullptr, nullptr, nullptr, B * MTOK, INNER, C);
  gemm_nt_kernel<<<dim3(INNER / 64, (B * MTOK) / 128), 128, 0, stream>>>(
      yn16, wv16, v16, nullptr, bv, nullptr, B * MTOK, INNER, C);

  attn_kernel<<<dim3(NTOK / 128, B * 8), 256, 0, stream>>>(q16, k16, v16, at16,
                                                           NTOK, MTOK);

  gemm_nt_kernel<<<dim3(C / 64, (B * NTOK) / 128), 128, 0, stream>>>(
      at16, wo16, nullptr, out, bo, xn32, B * NTOK, C, INNER);
}
